// CMAFMBlock_83227876262631
// MI455X (gfx1250) — matrix operand order repaired and checked by reading, not compiled, not run on hardware
//
#include <hip/hip_runtime.h>
#include <hip/hip_bf16.h>

// ---------------------------------------------------------------------------
// CMAFM block for MI455X (gfx1250): all GEMMs on v_wmma_f32_16x16x32_bf16.
// All WMMA operands staged K-contiguous so fragments load as b128 pairs.
// bf16 conversion via native hardware cvt; rcp/rsq via fast intrinsics.
// ---------------------------------------------------------------------------

typedef __attribute__((ext_vector_type(16))) __bf16         v16bf;
typedef __attribute__((ext_vector_type(8)))  float          v8f;
typedef __attribute__((ext_vector_type(4)))  float          v4f;
typedef __attribute__((ext_vector_type(4)))  unsigned short v4us;
typedef __attribute__((ext_vector_type(8)))  unsigned short v8us;
typedef __attribute__((ext_vector_type(16))) unsigned short v16us;

#define C_   128
#define W_   256
#define HW_  65536          // 256*256
#define NH_  4
#define HD_  32

// LDS row pitches (elements); all multiples of 8 shorts (16 B) for alignment,
// chosen so row-stride % 64 banks advances by 4 (no same-bank striding).
#define LX_  136            // kernel1 X tile rows [n][c]
#define LQ_  136            // kernel2 q/k/y rows  [t][c]
#define LV_  72             // kernel2 v rows      [c][t]
#define LP_  72             // kernel2 p rows      [t][t]
#define LS_  66             // kernel2 s rows (f32)
#define LB_  264            // kernel3 rows        [n][k], k<256

// bf16 weight workspace offsets (in ushorts), after 8*NE of activations
#define OFF_QKVW   0u       // 6 x 16384
#define OFF_PROJ   98304u   // 2 x 16384
#define OFF_GOPT   131072u  // 32768
#define OFF_GSAR   163840u  // 32768
#define OFF_FUSE   196608u  // 32768

// ---- bf16 helpers: native hardware conversion ------------------------------
__device__ __forceinline__ unsigned short f2us(float f) {
    return __builtin_bit_cast(unsigned short, (__bf16)f);   // v_cvt bf16 (RNE)
}
__device__ __forceinline__ float us2f(unsigned short s) {
    unsigned u = ((unsigned)s) << 16;
    return __builtin_bit_cast(float, u);
}
__device__ __forceinline__ float rcpf_(float x) { return __builtin_amdgcn_rcpf(x); }
__device__ __forceinline__ float rsqf_(float x) { return __builtin_amdgcn_rsqf(x); }
__device__ __forceinline__ float sigmoidf_(float x) {
    return rcpf_(1.0f + __expf(-x));
}

// ---- WMMA fragment builders ------------------------------------------------
// A (16x32): lane l -> row m = l&15; per lane the 16 elements are two
// contiguous 8-element runs at K = (l>>4)*8 and 16+(l>>4)*8. -> two b128 loads.
__device__ __forceinline__ v16bf afrag16(const unsigned short* rowp, int lane) {
    int kh = (lane >> 4) * 8;
    v8us lo = *(const v8us*)(rowp + kh);
    v8us hi = *(const v8us*)(rowp + 16 + kh);
    v16us t = __builtin_shufflevector(lo, hi, 0, 1, 2, 3, 4, 5, 6, 7,
                                      8, 9, 10, 11, 12, 13, 14, 15);
    return __builtin_bit_cast(v16bf, t);
}
// B (32x16): lane l -> col n = l&15; per lane the 16 elements are two
// contiguous 8-element runs at K = (l>>4)*8 and 16+(l>>4)*8.  rowp = &src[n][k0] where src
// is stored [n][K] (K-contiguous).  -> two b128 loads.
__device__ __forceinline__ v16bf bfrag16(const unsigned short* rowp, int lane) {
    const unsigned short* q = rowp + ((lane >> 4) << 3);
    v8us lo = *(const v8us*)(q);
    v8us hi = *(const v8us*)(q + 16);
    v16us t = __builtin_shufflevector(lo, hi, 0, 1, 2, 3, 4, 5, 6, 7,
                                      8, 9, 10, 11, 12, 13, 14, 15);
    return __builtin_bit_cast(v16bf, t);
}
__device__ __forceinline__ v8f wmma_bf16(v16bf a, v16bf b, v8f c) {
    return __builtin_amdgcn_wmma_f32_16x16x32_bf16(false, a, false, b, (short)0, c,
                                                   false, false);
}
__device__ __forceinline__ v8f zero8() {
    v8f z;
#pragma unroll
    for (int r = 0; r < 8; ++r) z[r] = 0.0f;
    return z;
}

// ===========================================================================
// Kernel 0: f32 -> bf16 weight conversion (one-shot prep, ~450 KB total).
// ===========================================================================
__global__ __launch_bounds__(256) void f32_to_bf16_kernel(
    const float* __restrict__ s, unsigned short* __restrict__ d, int n)
{
    int i = blockIdx.x * 256 + threadIdx.x;
    if (i < n) d[i] = f2us(s[i]);
}

// ===========================================================================
// Kernel 1: six 1x1 convs (Q/K/V both modalities) -> bf16 workspace.
// Grid: (HW/16, B, 6). Block 256 = 8 waves; wave w owns out-channel tile w.
// Weights read as bf16 from global (L2-resident); X tile staged [n][c] in LDS.
// ===========================================================================
__global__ __launch_bounds__(256) void qkv_conv_kernel(
    const float* __restrict__ F_opt, const float* __restrict__ F_sar,
    const unsigned short* __restrict__ wbf,     // 6 x (128x128) bf16
    const float* __restrict__ bq_opt, const float* __restrict__ bk_opt,
    const float* __restrict__ bv_opt, const float* __restrict__ bq_sar,
    const float* __restrict__ bk_sar, const float* __restrict__ bv_sar,
    unsigned short* __restrict__ qkv_ws, size_t NE)
{
    const int tid  = threadIdx.x;
    const int lane = tid & 31;
    const int wave = tid >> 5;
    const int p0   = blockIdx.x * 16;
    const int b    = blockIdx.y;
    const int z    = blockIdx.z;

    const float* X = (z < 3) ? F_opt : F_sar;
    const float* Bs;
    switch (z) {
        case 0:  Bs = bq_opt; break;
        case 1:  Bs = bk_opt; break;
        case 2:  Bs = bv_opt; break;
        case 3:  Bs = bq_sar; break;
        case 4:  Bs = bk_sar; break;
        default: Bs = bv_sar; break;
    }
    const unsigned short* Wt = wbf + (size_t)z * (C_ * C_);
    unsigned short* out = qkv_ws + (size_t)z * NE;

    __shared__ unsigned short x_lds[16 * LX_];   // [n][c] bf16, K-contiguous

    // 2048 floats, float4-vectorized: i -> c = i>>2, n0 = (i&3)*4
    for (int i = tid; i < 512; i += 256) {
        int c = i >> 2, n0 = (i & 3) << 2;
        v4f x = *(const v4f*)(X + ((size_t)(b * C_ + c)) * HW_ + p0 + n0);
#pragma unroll
        for (int j = 0; j < 4; ++j) x_lds[(n0 + j) * LX_ + c] = f2us(x[j]);
    }
    __syncthreads();

    const int Mbase = wave * 16;
    const int m = Mbase + (lane & 15);
    const int n = lane & 15;

    v8f acc = zero8();
#pragma unroll
    for (int kk = 0; kk < 4; ++kk) {
        v16bf a  = afrag16(Wt + m * C_ + kk * 32, lane);        // global b128
        v16bf bb = bfrag16(&x_lds[n * LX_ + kk * 32], lane);    // ds b128
        acc = wmma_bf16(a, bb, acc);
    }
#pragma unroll
    for (int r = 0; r < 8; ++r) {
        int mo = Mbase + r + ((lane >> 4) << 3);
        float val = acc[r] + Bs[mo];
        out[((size_t)(b * C_ + mo)) * HW_ + p0 + n] = f2us(val);
    }
}

// ===========================================================================
// Kernel 2: windowed cross-attention + projection.
// Grid: (nW=1024, B, dir=2). Block 128 = 4 waves; heads one at a time.
// q/k stored [t][c] (A and B both K-contiguous), v stored [c][t] so the
// P*V B-operand is K-contiguous too.
// ===========================================================================
__global__ __launch_bounds__(128) void window_attn_kernel(
    const unsigned short* __restrict__ qkv_ws,
    const unsigned short* __restrict__ wbf,
    const float* __restrict__ proj_b_o2s, const float* __restrict__ proj_b_s2o,
    unsigned short* __restrict__ attn_ws, size_t NE)
{
    const int tid  = threadIdx.x;
    const int lane = tid & 31;
    const int wave = tid >> 5;
    const int win  = blockIdx.x;                 // 0..1023
    const int b    = blockIdx.y;
    const int dir  = blockIdx.z;                 // 0: o2s, 1: s2o
    const int wy   = win >> 5, wx = win & 31;
    const int p0   = wy * 8 * W_ + wx * 8;

    const unsigned short* Qsrc = qkv_ws + (size_t)(dir ? 3 : 0) * NE;
    const unsigned short* Ksrc = qkv_ws + (size_t)(dir ? 1 : 4) * NE;
    const unsigned short* Vsrc = qkv_ws + (size_t)(dir ? 2 : 5) * NE;
    const unsigned short* PwBf = wbf + OFF_PROJ + (size_t)dir * (C_ * C_);
    const float* Pb = dir ? proj_b_s2o : proj_b_o2s;
    unsigned short* Out = attn_ws + (size_t)dir * NE;

    __shared__ unsigned short q_lds[64 * LQ_];   // [t][c]
    __shared__ unsigned short k_lds[64 * LQ_];   // [t][c]
    __shared__ unsigned short v_lds[C_ * LV_];   // [c][t]  (transposed)
    __shared__ float          s_lds[64 * LS_];   // logits, one head
    __shared__ unsigned short p_lds[64 * LP_];   // probs bf16 [t][t]
    __shared__ unsigned short y_lds[64 * LQ_];   // concat head outputs [t][c]

    // load window: 128 ch x 64 tokens, 4-short vector loads (one 8x8 row half)
    for (int i = tid; i < C_ * 16; i += 128) {
        int c = i >> 4, quad = i & 15;
        int t0 = quad * 4;
        int p = p0 + (t0 >> 3) * W_ + (t0 & 7);
        size_t src = ((size_t)(b * C_ + c)) * HW_ + p;
        v4us qv = *(const v4us*)(Qsrc + src);
        v4us kv = *(const v4us*)(Ksrc + src);
        v4us vv = *(const v4us*)(Vsrc + src);
#pragma unroll
        for (int j = 0; j < 4; ++j) {
            int t = t0 + j;
            q_lds[t * LQ_ + c] = qv[j];
            k_lds[t * LQ_ + c] = kv[j];
            v_lds[c * LV_ + t] = vv[j];
        }
    }
    __syncthreads();

    const float scale = 0.17677669529663687f;    // 32^-0.5

    for (int h = 0; h < NH_; ++h) {
        const int ch0 = h * HD_;
        // ---- S = Qh * Kh^T  (64x64, K=32): 16 tiles, 4 per wave ----
#pragma unroll
        for (int i = 0; i < 4; ++i) {
            int id = wave * 4 + i, mt = id >> 2, nt = id & 3;
            int m = mt * 16 + (lane & 15);
            int n = nt * 16 + (lane & 15);
            v8f acc = zero8();
            v16bf a  = afrag16(&q_lds[m * LQ_ + ch0], lane);
            v16bf bb = bfrag16(&k_lds[n * LQ_ + ch0], lane);  // B[k][n]=Kh[n][k]
            acc = wmma_bf16(a, bb, acc);
#pragma unroll
            for (int r = 0; r < 8; ++r) {
                int m2 = mt * 16 + r + ((lane >> 4) << 3);
                s_lds[m2 * LS_ + (nt * 16 + (lane & 15))] = acc[r] * scale;
            }
        }
        __syncthreads();
        // ---- row softmax (64 rows, threads 0..63) ----
        if (tid < 64) {
            float mx = -1e30f;
            for (int j = 0; j < 64; ++j) mx = fmaxf(mx, s_lds[tid * LS_ + j]);
            float sum = 0.0f;
            for (int j = 0; j < 64; ++j) {
                float e = __expf(s_lds[tid * LS_ + j] - mx);
                s_lds[tid * LS_ + j] = e;
                sum += e;
            }
            float inv = rcpf_(sum);
            for (int j = 0; j < 64; ++j)
                p_lds[tid * LP_ + j] = f2us(s_lds[tid * LS_ + j] * inv);
        }
        __syncthreads();
        // ---- O = P * Vh (64x32, K=64): 8 tiles, 2 per wave ----
#pragma unroll
        for (int i = 0; i < 2; ++i) {
            int id = wave * 2 + i, mt = id >> 1, nt = id & 1;
            int m  = mt * 16 + (lane & 15);
            int nh = nt * 16 + (lane & 15);
            v8f acc = zero8();
#pragma unroll
            for (int kk = 0; kk < 2; ++kk) {
                v16bf a  = afrag16(&p_lds[m * LP_ + kk * 32], lane);
                v16bf bb = bfrag16(&v_lds[(ch0 + nh) * LV_ + kk * 32], lane);
                acc = wmma_bf16(a, bb, acc);
            }
#pragma unroll
            for (int r = 0; r < 8; ++r) {
                int m2 = mt * 16 + r + ((lane >> 4) << 3);
                y_lds[m2 * LQ_ + ch0 + nh] = f2us(acc[r]);
            }
        }
        __syncthreads();
    }

    // ---- projection: Out(64x128) = Y(64x128) * Pw^T, K=128 ----
    const int mt = wave;                          // wave owns 16 tokens
#pragma unroll
    for (int ot = 0; ot < 8; ++ot) {
        int m = mt * 16 + (lane & 15);
        int o = ot * 16 + (lane & 15);
        v8f acc = zero8();
#pragma unroll
        for (int kk = 0; kk < 4; ++kk) {
            v16bf a  = afrag16(&y_lds[m * LQ_ + kk * 32], lane);
            v16bf bb = bfrag16(PwBf + o * C_ + kk * 32, lane);  // B[k][o]=Pw[o][k]
            acc = wmma_bf16(a, bb, acc);
        }
#pragma unroll
        for (int r = 0; r < 8; ++r) {
            int m2 = mt * 16 + r + ((lane >> 4) << 3);
            float val = acc[r] + Pb[o];
            int p = p0 + (m2 >> 3) * W_ + (m2 & 7);
            Out[((size_t)(b * C_ + o)) * HW_ + p] = f2us(val);
        }
    }
}

// ===========================================================================
// Kernel 3: gates (K=256), residual blend, fuse (K=256), BN + SiLU.
// Grid: (HW/16, B). Block 256 = 8 waves; wave w owns out-channel tile w.
// B operands stored [n][k] in LDS (K-contiguous).
// d_out layout: [0,NE) fused | [NE,2NE) sigma_opt | [2NE,3NE) sigma_sar.
// ===========================================================================
__global__ __launch_bounds__(256) void fuse_kernel(
    const float* __restrict__ F_opt, const float* __restrict__ F_sar,
    const unsigned short* __restrict__ f_o2s, const unsigned short* __restrict__ f_s2o,
    const unsigned short* __restrict__ wbf,
    const float* __restrict__ gate_opt_b, const float* __restrict__ gate_sar_b,
    const float* __restrict__ fuse_b,
    const float* __restrict__ bn_gamma, const float* __restrict__ bn_beta,
    const float* __restrict__ bn_mean, const float* __restrict__ bn_var,
    float* __restrict__ out, size_t NE)
{
    const int tid  = threadIdx.x;
    const int lane = tid & 31;
    const int wave = tid >> 5;
    const int p0   = blockIdx.x * 16;
    const int b    = blockIdx.y;

    const unsigned short* GwOpt = wbf + OFF_GOPT;
    const unsigned short* GwSar = wbf + OFF_GSAR;
    const unsigned short* FwBf  = wbf + OFF_FUSE;

    __shared__ unsigned short bopt[16 * LB_];    // [n][F_opt(128); F_o2s(128)]
    __shared__ unsigned short bsar[16 * LB_];    // [n][F_sar(128); F_s2o(128)]
    __shared__ unsigned short fnew[16 * LB_];    // [n][F_opt_new ; F_sar_new]

    // f32 halves (c<128): float4 loads; bf16 halves: 4-short loads
    for (int i = tid; i < 512; i += 256) {
        int c = i >> 2, n0 = (i & 3) << 2;
        size_t src = ((size_t)(b * C_ + c)) * HW_ + p0 + n0;
        v4f  xo = *(const v4f*)(F_opt + src);
        v4f  xs = *(const v4f*)(F_sar + src);
        v4us ao = *(const v4us*)(f_o2s + src);
        v4us as = *(const v4us*)(f_s2o + src);
#pragma unroll
        for (int j = 0; j < 4; ++j) {
            bopt[(n0 + j) * LB_ + c]       = f2us(xo[j]);
            bsar[(n0 + j) * LB_ + c]       = f2us(xs[j]);
            bopt[(n0 + j) * LB_ + C_ + c]  = ao[j];
            bsar[(n0 + j) * LB_ + C_ + c]  = as[j];
        }
    }
    __syncthreads();

    const int Mbase = wave * 16;
    const int m = Mbase + (lane & 15);
    const int n = lane & 15;

    // ---- gate GEMMs + sigmoid + residual blend ----
    for (int which = 0; which < 2; ++which) {
        const unsigned short* Gw = which ? GwSar : GwOpt;
        const float* Gb = which ? gate_sar_b : gate_opt_b;
        const unsigned short* Bm = which ? bsar : bopt;
        const float* Fx = which ? F_sar : F_opt;
        const unsigned short* Fa = which ? f_s2o : f_o2s;

        v8f acc = zero8();
#pragma unroll
        for (int kk = 0; kk < 8; ++kk) {
            v16bf a  = afrag16(Gw + m * 256 + kk * 32, lane);
            v16bf bb = bfrag16(&Bm[n * LB_ + kk * 32], lane);
            acc = wmma_bf16(a, bb, acc);
        }
#pragma unroll
        for (int r = 0; r < 8; ++r) {
            int mo = Mbase + r + ((lane >> 4) << 3);
            size_t idx = ((size_t)(b * C_ + mo)) * HW_ + p0 + n;
            float sig = sigmoidf_(acc[r] + Gb[mo]);
            out[NE * (size_t)(1 + which) + idx] = sig;                 // sigma
            float blended = Fx[idx] + sig * us2f(Fa[idx]);             // residual
            fnew[n * LB_ + which * C_ + mo] = f2us(blended);
        }
    }
    __syncthreads();

    // ---- fuse GEMM + BatchNorm + SiLU ----
    v8f acc = zero8();
#pragma unroll
    for (int kk = 0; kk < 8; ++kk) {
        v16bf a  = afrag16(FwBf + m * 256 + kk * 32, lane);
        v16bf bb = bfrag16(&fnew[n * LB_ + kk * 32], lane);
        acc = wmma_bf16(a, bb, acc);
    }
#pragma unroll
    for (int r = 0; r < 8; ++r) {
        int mo = Mbase + r + ((lane >> 4) << 3);
        float val = acc[r] + fuse_b[mo];
        float inv = bn_gamma[mo] * rsqf_(bn_var[mo] + 1e-5f);
        val = val * inv + (bn_beta[mo] - bn_mean[mo] * inv);
        out[((size_t)(b * C_ + mo)) * HW_ + p0 + n] = val * sigmoidf_(val);
    }
}

// ===========================================================================
extern "C" void kernel_launch(void* const* d_in, const int* in_sizes, int n_in,
                              void* d_out, int out_size, void* d_ws, size_t ws_size,
                              hipStream_t stream) {
    (void)n_in; (void)out_size; (void)ws_size;
    const float* F_opt      = (const float*)d_in[0];
    const float* F_sar      = (const float*)d_in[1];
    const float* wq_opt     = (const float*)d_in[2];
    const float* bq_opt     = (const float*)d_in[3];
    const float* wk_opt     = (const float*)d_in[4];
    const float* bk_opt     = (const float*)d_in[5];
    const float* wv_opt     = (const float*)d_in[6];
    const float* bv_opt     = (const float*)d_in[7];
    const float* wq_sar     = (const float*)d_in[8];
    const float* bq_sar     = (const float*)d_in[9];
    const float* wk_sar     = (const float*)d_in[10];
    const float* bk_sar     = (const float*)d_in[11];
    const float* wv_sar     = (const float*)d_in[12];
    const float* bv_sar     = (const float*)d_in[13];
    const float* proj_w_o2s = (const float*)d_in[14];
    const float* proj_b_o2s = (const float*)d_in[15];
    const float* proj_w_s2o = (const float*)d_in[16];
    const float* proj_b_s2o = (const float*)d_in[17];
    const float* gate_opt_w = (const float*)d_in[18];
    const float* gate_opt_b = (const float*)d_in[19];
    const float* gate_sar_w = (const float*)d_in[20];
    const float* gate_sar_b = (const float*)d_in[21];
    const float* fuse_w     = (const float*)d_in[22];
    const float* fuse_b     = (const float*)d_in[23];
    const float* bn_gamma   = (const float*)d_in[24];
    const float* bn_beta    = (const float*)d_in[25];
    const float* bn_mean    = (const float*)d_in[26];
    const float* bn_var     = (const float*)d_in[27];

    const int B = in_sizes[0] / (C_ * HW_);          // 4
    const size_t NE = (size_t)B * C_ * HW_;          // elems per (B,C,H,W) tensor

    // Workspace (ushort): [0,6NE) Q/K/V, [6NE,8NE) attn outs, then bf16 weights.
    unsigned short* qkv_ws  = (unsigned short*)d_ws;
    unsigned short* attn_ws = qkv_ws + (size_t)6 * NE;
    unsigned short* wbf     = qkv_ws + (size_t)8 * NE;

    // ---- weight conversion (f32 -> bf16) ----
    const int NW = C_ * C_;                          // 16384
    f32_to_bf16_kernel<<<NW / 256, 256, 0, stream>>>(wq_opt, wbf + 0 * NW, NW);
    f32_to_bf16_kernel<<<NW / 256, 256, 0, stream>>>(wk_opt, wbf + 1 * NW, NW);
    f32_to_bf16_kernel<<<NW / 256, 256, 0, stream>>>(wv_opt, wbf + 2 * NW, NW);
    f32_to_bf16_kernel<<<NW / 256, 256, 0, stream>>>(wq_sar, wbf + 3 * NW, NW);
    f32_to_bf16_kernel<<<NW / 256, 256, 0, stream>>>(wk_sar, wbf + 4 * NW, NW);
    f32_to_bf16_kernel<<<NW / 256, 256, 0, stream>>>(wv_sar, wbf + 5 * NW, NW);
    f32_to_bf16_kernel<<<NW / 256, 256, 0, stream>>>(proj_w_o2s, wbf + OFF_PROJ, NW);
    f32_to_bf16_kernel<<<NW / 256, 256, 0, stream>>>(proj_w_s2o, wbf + OFF_PROJ + NW, NW);
    f32_to_bf16_kernel<<<2 * NW / 256, 256, 0, stream>>>(gate_opt_w, wbf + OFF_GOPT, 2 * NW);
    f32_to_bf16_kernel<<<2 * NW / 256, 256, 0, stream>>>(gate_sar_w, wbf + OFF_GSAR, 2 * NW);
    f32_to_bf16_kernel<<<2 * NW / 256, 256, 0, stream>>>(fuse_w, wbf + OFF_FUSE, 2 * NW);

    // ---- pipeline ----
    dim3 g1(HW_ / 16, B, 6);
    qkv_conv_kernel<<<g1, 256, 0, stream>>>(
        F_opt, F_sar, wbf, bq_opt, bk_opt, bv_opt, bq_sar, bk_sar, bv_sar,
        qkv_ws, NE);

    dim3 g2((W_ / 8) * (W_ / 8), B, 2);
    window_attn_kernel<<<g2, 128, 0, stream>>>(
        qkv_ws, wbf, proj_b_o2s, proj_b_s2o, attn_ws, NE);

    dim3 g3(HW_ / 16, B);
    fuse_kernel<<<g3, 256, 0, stream>>>(
        F_opt, F_sar, attn_ws, attn_ws + NE, wbf,
        gate_opt_b, gate_sar_b, fuse_b,
        bn_gamma, bn_beta, bn_mean, bn_var,
        (float*)d_out, NE);
}
